// TriangularSylvesterVAE_PointTransformer_2731599200754
// MI455X (gfx1250) — matrix operand order repaired and checked by reading, not compiled, not run on hardware
//
#include <hip/hip_runtime.h>
#include <math.h>

#define NN 50000
#define EE 400000
#define NEDGE (EE + NN)
#define HEADS 5
#define KFLOWS 6
#define NEG_SLOPE 0.2f
#define BN_EPS 1e-5f
#define ORD_NEG_INF 0x007FFFFFu  /* order-preserving encoding of -inf */
#define KMAX_PAD 192             /* max padded K over all layers (160 -> 160, pad room) */
#define WPB 8                    /* waves per block in the GEMM */

typedef __attribute__((ext_vector_type(16))) __bf16 v16bf;
typedef __attribute__((ext_vector_type(8)))  float  v8f;

// ---------------- helpers ----------------

__device__ __forceinline__ unsigned short f2bfu(float f) {  // f32 -> bf16 bits, RNE
  union { float f; unsigned u; } a; a.f = f;
  return (unsigned short)((a.u + 0x7fffu + ((a.u >> 16) & 1u)) >> 16);
}

union BF16x16 {
  v16bf v;
  unsigned short s[16];
  unsigned u[8];
};

__device__ __forceinline__ unsigned f2ord(float f) {
  unsigned u = __float_as_uint(f);
  return (u & 0x80000000u) ? ~u : (u | 0x80000000u);
}
__device__ __forceinline__ float ord2f(unsigned o) {
  return (o & 0x80000000u) ? __uint_as_float(o ^ 0x80000000u) : __uint_as_float(~o);
}

// ---------------- generic WMMA GEMM: C = act(A[M,K] * W[K,Nc] + bias) ----------------
// grid.y = column tile, grid.x covers row tiles (WPB waves/block, 1 tile/wave).
// The whole block shares one 16-column weight panel staged in LDS (bf16, transposed,
// zero-padded to a multiple of 32 in K). A-panel loads are 4x b128 per K-chunk;
// the K-tail uses branchless masked scalar loads (unconditionally consumed).

template <int ACT>  // 0 = none, 1 = relu, 2 = tanh
__global__ void gemm_wmma_k(const float* __restrict__ A, const float* __restrict__ W,
                            const float* __restrict__ bias, float* __restrict__ C,
                            int M, int K, int Nc, int mtiles) {
  __shared__ __align__(16) unsigned short wlds[16 * KMAX_PAD];  // [n][k] transposed panel

  const int col0 = blockIdx.y * 16;
  const int kpad = (K + 31) & ~31;

  // cooperative, coalesced W panel load + bf16 convert + transpose into LDS
  for (int idx = threadIdx.x; idx < 16 * kpad; idx += blockDim.x) {
    int n = idx & 15;
    int k = idx >> 4;
    int cg = col0 + n;
    float v = (k < K && cg < Nc) ? W[(size_t)k * (size_t)Nc + cg] : 0.0f;
    wlds[n * KMAX_PAD + k] = f2bfu(v);
  }
  __syncthreads();

  const int wave = threadIdx.x >> 5;
  const int mt = blockIdx.x * WPB + wave;
  if (mt >= mtiles) return;                 // wave-uniform; EXEC all-ones for WMMA

  const int lane = threadIdx.x & 31;
  const int hi = lane >> 4;
  const int lr = lane & 15;
  const int off = hi * 8;                   // A-fragment K sub-offset for hi lanes
  const int arow = mt * 16 + lr;
  const int col = col0 + lr;
  const bool colOK = (col < Nc);

  const int arow_c = (arow < M) ? arow : (M - 1);   // clamp: OOB rows never stored
  const float* __restrict__ Arow = A + (size_t)arow_c * (size_t)K;
  const unsigned* __restrict__ bbase =
      (const unsigned*)(wlds + lr * KMAX_PAD);      // this lane's B column (bf16 pairs)

  float bv = (colOK && bias) ? bias[col] : 0.0f;
  v8f acc;
#pragma unroll
  for (int r = 0; r < 8; ++r) acc[r] = bv;

  const int kfull = K & ~31;
  for (int k0 = 0; k0 < kfull; k0 += 32) {
    __builtin_prefetch(Arow + k0 + 32 + off, 0, 1);
    BF16x16 af, bf;
    // A: two contiguous 8-float runs -> 4x float4 (global_load_b128)
    const float4* p0 = (const float4*)(Arow + k0 + off);
    const float4* p1 = (const float4*)(Arow + k0 + 16 + off);
    float4 a0 = p0[0], a1 = p0[1];
    float4 a2 = p1[0], a3 = p1[1];
    af.s[0] = f2bfu(a0.x); af.s[1] = f2bfu(a0.y); af.s[2] = f2bfu(a0.z); af.s[3] = f2bfu(a0.w);
    af.s[4] = f2bfu(a1.x); af.s[5] = f2bfu(a1.y); af.s[6] = f2bfu(a1.z); af.s[7] = f2bfu(a1.w);
    af.s[8] = f2bfu(a2.x); af.s[9] = f2bfu(a2.y); af.s[10] = f2bfu(a2.z); af.s[11] = f2bfu(a2.w);
    af.s[12] = f2bfu(a3.x); af.s[13] = f2bfu(a3.y); af.s[14] = f2bfu(a3.z); af.s[15] = f2bfu(a3.w);
    // B: two 8-bf16 runs from LDS at k0+8h and k0+16+8h (matches A)
    const unsigned* bp = bbase + ((k0 + hi * 8) >> 1);
#pragma unroll
    for (int j = 0; j < 8; ++j) bf.u[j] = bp[(j < 4) ? j : (j + 4)];
    acc = __builtin_amdgcn_wmma_f32_16x16x32_bf16(false, af.v, false, bf.v, (short)0, acc,
                                                  false, false);
  }
  if (kfull < K) {  // K tail: branchless clamped loads, result always consumed (AND mask)
    BF16x16 af, bf;
#pragma unroll
    for (int e = 0; e < 16; ++e) {
      int ka = kfull + ((e < 8) ? e : (e + 8)) + off;
      int kc = (ka < K) ? ka : (K - 1);
      unsigned msk = (ka < K) ? 0xffffu : 0u;
      af.s[e] = (unsigned short)(f2bfu(Arow[kc]) & msk);
    }
    const unsigned* bp = bbase + ((kfull + hi * 8) >> 1);
#pragma unroll
    for (int j = 0; j < 8; ++j) bf.u[j] = bp[(j < 4) ? j : (j + 4)];
    acc = __builtin_amdgcn_wmma_f32_16x16x32_bf16(false, af.v, false, bf.v, (short)0, acc,
                                                  false, false);
  }

  // store epilogue: one exec mask for colOK, uniform full-tile fast path for M
  if (colOK) {
    float* __restrict__ cp = C + (size_t)(mt * 16 + (hi ? 8 : 0)) * (size_t)Nc + col;
    if (mt * 16 + 16 <= M) {
#pragma unroll
      for (int r = 0; r < 8; ++r) {
        float v = acc[r];
        if (ACT == 1) v = fmaxf(v, 0.0f);
        if (ACT == 2) v = tanhf(v);
        cp[(size_t)r * (size_t)Nc] = v;
      }
    } else {
#pragma unroll
      for (int r = 0; r < 8; ++r) {
        int m = mt * 16 + r + (hi ? 8 : 0);
        if (m < M) {
          float v = acc[r];
          if (ACT == 1) v = fmaxf(v, 0.0f);
          if (ACT == 2) v = tanhf(v);
          cp[(size_t)r * (size_t)Nc] = v;
        }
      }
    }
  }
}

// ---------------- utility kernels ----------------

__global__ void fill_f32_k(float* __restrict__ p, float v, long long n) {
  long long i = (long long)blockIdx.x * blockDim.x + threadIdx.x;
  if (i < n) p[i] = v;
}
__global__ void fill_u32_k(unsigned* __restrict__ p, unsigned v, long long n) {
  long long i = (long long)blockIdx.x * blockDim.x + threadIdx.x;
  if (i < n) p[i] = v;
}

__global__ void build_edges_k(const int* __restrict__ ei, int* __restrict__ src,
                              int* __restrict__ dst) {
  int i = blockIdx.x * blockDim.x + threadIdx.x;
  if (i >= NEDGE) return;
  if (i < EE) { src[i] = ei[i]; dst[i] = ei[EE + i]; }
  else        { src[i] = i - EE; dst[i] = i - EE; }
}

// deterministic per-channel stats: one block per channel
__global__ void colstats_k(const float* __restrict__ X, int rows, int C,
                           float* __restrict__ sums, float* __restrict__ sumsq) {
  __shared__ float ss[256], qq[256];
  int c = blockIdx.x;
  float s = 0.f, q = 0.f;
  for (int r = threadIdx.x; r < rows; r += 256) {
    float v = X[(size_t)r * (size_t)C + c];
    s += v; q += v * v;
  }
  ss[threadIdx.x] = s; qq[threadIdx.x] = q;
  __syncthreads();
  for (int o = 128; o > 0; o >>= 1) {
    if ((int)threadIdx.x < o) { ss[threadIdx.x] += ss[threadIdx.x + o];
                                qq[threadIdx.x] += qq[threadIdx.x + o]; }
    __syncthreads();
  }
  if (threadIdx.x == 0) { sums[c] = ss[0]; sumsq[c] = qq[0]; }
}

__global__ void bn_params_k(const float* __restrict__ sums, const float* __restrict__ sumsq,
                            float invCnt, int C, float* __restrict__ mean,
                            float* __restrict__ istd) {
  int c = blockIdx.x * blockDim.x + threadIdx.x;
  if (c < C) {
    float m = sums[c] * invCnt;
    float v = sumsq[c] * invCnt - m * m;
    mean[c] = m;
    istd[c] = rsqrtf(v + BN_EPS);
  }
}

__global__ void bn_relu_k(float* __restrict__ X, const float* __restrict__ mean,
                          const float* __restrict__ istd, long long total, int C) {
  long long i = (long long)blockIdx.x * blockDim.x + threadIdx.x;
  if (i >= total) return;
  int c = (int)(i % C);
  float v = (X[i] - mean[c]) * istd[c];
  X[i] = fmaxf(v, 0.f);
}

// ---------------- Point-Transformer edge kernels ----------------

__global__ void pt_delta_raw_k(const float* __restrict__ x, const int* __restrict__ src,
                               const int* __restrict__ dst, const float* __restrict__ pw,
                               const float* __restrict__ pb, float* __restrict__ out, int C) {
  long long i = (long long)blockIdx.x * blockDim.x + threadIdx.x;
  long long tot = (long long)NEDGE * C;
  if (i >= tot) return;
  int e = (int)(i / C), c = (int)(i % C);
  int s = src[e], d = dst[e];
  float r0 = x[(size_t)d * 7 + 5] - x[(size_t)s * 7 + 5];
  float r1 = x[(size_t)d * 7 + 6] - x[(size_t)s * 7 + 6];
  out[i] = r0 * pw[c] + r1 * pw[C + c] + pb[c];
}

// eb0: draw -> alpha_pre (in place), delta = relu(bn(draw))
__global__ void pt_alpha_pre_k(float* __restrict__ eb0, const float* __restrict__ asrc,
                               const float* __restrict__ adst, const int* __restrict__ src,
                               const int* __restrict__ dst, const float* __restrict__ m0,
                               const float* __restrict__ i0, int C) {
  long long i = (long long)blockIdx.x * blockDim.x + threadIdx.x;
  long long tot = (long long)NEDGE * C;
  if (i >= tot) return;
  int e = (int)(i / C), c = (int)(i % C);
  float dl = fmaxf((eb0[i] - m0[c]) * i0[c], 0.f);
  eb0[i] = adst[(size_t)dst[e] * C + c] - asrc[(size_t)src[e] * C + c] + dl;
}

__global__ void pt_segmax_k(const float* __restrict__ a, const int* __restrict__ dst,
                            unsigned* __restrict__ segU, int C) {
  long long i = (long long)blockIdx.x * blockDim.x + threadIdx.x;
  long long tot = (long long)NEDGE * C;
  if (i >= tot) return;
  int e = (int)(i / C), c = (int)(i % C);
  atomicMax(&segU[(size_t)dst[e] * C + c], f2ord(a[i]));
}

__global__ void pt_segsum_k(const float* __restrict__ a, const int* __restrict__ dst,
                            const unsigned* __restrict__ segU, float* __restrict__ segsum,
                            int C) {
  long long i = (long long)blockIdx.x * blockDim.x + threadIdx.x;
  long long tot = (long long)NEDGE * C;
  if (i >= tot) return;
  int e = (int)(i / C), c = (int)(i % C);
  float m = ord2f(segU[(size_t)dst[e] * C + c]);
  atomicAdd(&segsum[(size_t)dst[e] * C + c], __expf(a[i] - m));
}

__global__ void pt_scatter_k(const float* __restrict__ a, const float* __restrict__ x,
                             const int* __restrict__ src, const int* __restrict__ dst,
                             const float* __restrict__ pw, const float* __restrict__ pb,
                             const float* __restrict__ m0, const float* __restrict__ i0,
                             const unsigned* __restrict__ segU, const float* __restrict__ segsum,
                             const float* __restrict__ val, float* __restrict__ outacc, int C) {
  long long i = (long long)blockIdx.x * blockDim.x + threadIdx.x;
  long long tot = (long long)NEDGE * C;
  if (i >= tot) return;
  int e = (int)(i / C), c = (int)(i % C);
  int s = src[e], d = dst[e];
  float r0 = x[(size_t)d * 7 + 5] - x[(size_t)s * 7 + 5];
  float r1 = x[(size_t)d * 7 + 6] - x[(size_t)s * 7 + 6];
  float dl = fmaxf((r0 * pw[c] + r1 * pw[C + c] + pb[c] - m0[c]) * i0[c], 0.f);
  float mx = ord2f(segU[(size_t)d * C + c]);
  float alpha = __expf(a[i] - mx) / (segsum[(size_t)d * C + c] + 1e-16f);
  atomicAdd(&outacc[(size_t)d * C + c], alpha * (val[(size_t)s * C + c] + dl));
}

// ---------------- GAT edge kernels ----------------

__global__ void gat_logits_k(const float* __restrict__ xl, const float* __restrict__ xr,
                             const float* __restrict__ att, const int* __restrict__ src,
                             const int* __restrict__ dst, float* __restrict__ logits, int C) {
  long long i = (long long)blockIdx.x * blockDim.x + threadIdx.x;
  long long tot = (long long)NEDGE * HEADS;
  if (i >= tot) return;
  int e = (int)(i / HEADS), h = (int)(i % HEADS);
  const float* pl = xl + (size_t)src[e] * HEADS * C + (size_t)h * C;
  const float* pr = xr + (size_t)dst[e] * HEADS * C + (size_t)h * C;
  const float* pa = att + (size_t)h * C;
  float acc = 0.f;
  for (int c = 0; c < C; ++c) {
    float v = pl[c] + pr[c];
    v = (v >= 0.f) ? v : NEG_SLOPE * v;
    acc += v * pa[c];
  }
  logits[i] = acc;
}

__global__ void gat_segmax_k(const float* __restrict__ lg, const int* __restrict__ dst,
                             unsigned* __restrict__ segU) {
  long long i = (long long)blockIdx.x * blockDim.x + threadIdx.x;
  long long tot = (long long)NEDGE * HEADS;
  if (i >= tot) return;
  int e = (int)(i / HEADS), h = (int)(i % HEADS);
  atomicMax(&segU[(size_t)dst[e] * HEADS + h], f2ord(lg[i]));
}

__global__ void gat_segsum_k(const float* __restrict__ lg, const int* __restrict__ dst,
                             const unsigned* __restrict__ segU, float* __restrict__ segsum) {
  long long i = (long long)blockIdx.x * blockDim.x + threadIdx.x;
  long long tot = (long long)NEDGE * HEADS;
  if (i >= tot) return;
  int e = (int)(i / HEADS), h = (int)(i % HEADS);
  float m = ord2f(segU[(size_t)dst[e] * HEADS + h]);
  atomicAdd(&segsum[(size_t)dst[e] * HEADS + h], __expf(lg[i] - m));
}

__global__ void gat_scatter_k(const float* __restrict__ lg, const float* __restrict__ xl,
                              const int* __restrict__ src, const int* __restrict__ dst,
                              const unsigned* __restrict__ segU, const float* __restrict__ segsum,
                              float* __restrict__ outacc, int C) {
  long long i = (long long)blockIdx.x * blockDim.x + threadIdx.x;
  long long tot = (long long)NEDGE * HEADS;
  if (i >= tot) return;
  int e = (int)(i / HEADS), h = (int)(i % HEADS);
  int s = src[e], d = dst[e];
  float mx = ord2f(segU[(size_t)d * HEADS + h]);
  float alpha = __expf(lg[i] - mx) / (segsum[(size_t)d * HEADS + h] + 1e-16f);
  const float* pl = xl + (size_t)s * HEADS * C + (size_t)h * C;
  float* po = outacc + (size_t)d * HEADS * C + (size_t)h * C;
  for (int c = 0; c < C; ++c) atomicAdd(&po[c], alpha * pl[c]);
}

__global__ void gat_finish_concat_k(const float* __restrict__ outacc,
                                    const float* __restrict__ bias, float* __restrict__ y,
                                    int HC) {
  long long i = (long long)blockIdx.x * blockDim.x + threadIdx.x;
  long long tot = (long long)NN * HC;
  if (i >= tot) return;
  y[i] = outacc[i] + bias[(int)(i % HC)];
}

__global__ void gat_finish_mean_k(const float* __restrict__ outacc,
                                  const float* __restrict__ bias, float* __restrict__ y, int C) {
  long long i = (long long)blockIdx.x * blockDim.x + threadIdx.x;
  long long tot = (long long)NN * C;
  if (i >= tot) return;
  int n = (int)(i / C), c = (int)(i % C);
  float s = 0.f;
  for (int h = 0; h < HEADS; ++h) s += outacc[(size_t)n * HEADS * C + (size_t)h * C + c];
  y[i] = s * (1.0f / HEADS) + bias[c];
}

// ---------------- Sylvester flow head ----------------

__global__ void head_flow_k(const float* __restrict__ mu, const float* __restrict__ lv,
                            const float* __restrict__ eps, const float* __restrict__ dflat,
                            const float* __restrict__ dg1, const float* __restrict__ dg2,
                            const float* __restrict__ bflat, float* __restrict__ z0o,
                            float* __restrict__ zo, float* __restrict__ ldjo) {
  int n = blockIdx.x * blockDim.x + threadIdx.x;
  if (n >= NN) return;
  float z0[2], z[2];
#pragma unroll
  for (int i = 0; i < 2; ++i) {
    z0[i] = mu[(size_t)n * 2 + i] + eps[(size_t)n * 2 + i] * __expf(0.5f * lv[(size_t)n * 2 + i]);
    z[i] = z0[i];
  }
  float ldj = 0.f;
#pragma unroll
  for (int k = 0; k < KFLOWS; ++k) {
    float D01 = dflat[(size_t)n * 24 + 1 * KFLOWS + k];           // full_d[0][1]
    float D10 = dflat[(size_t)n * 24 + 2 * KFLOWS + k];           // full_d[1][0]
    float a0 = dg1[(size_t)n * 12 + k];                            // diag1[0] (tanh'd)
    float a1 = dg1[(size_t)n * 12 + KFLOWS + k];                   // diag1[1]
    float c0 = dg2[(size_t)n * 12 + k];
    float c1 = dg2[(size_t)n * 12 + KFLOWS + k];
    float b0 = bflat[(size_t)n * 12 + k];
    float b1 = bflat[(size_t)n * 12 + KFLOWS + k];
    float zp0 = (k & 1) ? z[1] : z[0];
    float zp1 = (k & 1) ? z[0] : z[1];
    float pre0 = zp0 * c0 + zp1 * D10 + b0;                        // r2 upper-tri
    float pre1 = zp1 * c1 + b1;
    float h0 = tanhf(pre0), h1 = tanhf(pre1);
    float zt0 = h0 * a0 + h1 * D01;                                // r1 upper-tri
    float zt1 = h1 * a1;
    if (k & 1) { float t = zt0; zt0 = zt1; zt1 = t; }
    z[0] += zt0; z[1] += zt1;
    float j0 = 1.f + (1.f - h0 * h0) * a0 * c0;
    float j1 = 1.f + (1.f - h1 * h1) * a1 * c1;
    ldj += __logf(fabsf(j0)) + __logf(fabsf(j1));
  }
  z0o[(size_t)n * 2] = z0[0]; z0o[(size_t)n * 2 + 1] = z0[1];
  zo[(size_t)n * 2] = z[0];   zo[(size_t)n * 2 + 1] = z[1];
  ldjo[n] = ldj;
}

// ---------------- host orchestration ----------------

struct GatP { const float *att, *bias, *bl, *br, *wl, *wr; };
struct PtP  { const float *attn_b, *attn_w, *dst_w, *lin_in_b, *lin_in_w,
                          *lin_out_b, *lin_out_w, *lin_w, *pos_b, *pos_w, *src_w; };

extern "C" void kernel_launch(void* const* d_in, const int* in_sizes, int n_in,
                              void* d_out, int out_size, void* d_ws, size_t ws_size,
                              hipStream_t stream) {
  (void)in_sizes; (void)n_in; (void)out_size; (void)ws_size;

  const float* x   = (const float*)d_in[0];
  const int*   ei  = (const int*)d_in[1];
  const float* eps = (const float*)d_in[2];

  // jax pytree (sorted dict keys) flattening: gdec[0..2], genc[0..1], head, pt[0..2]
  int p = 3;
  GatP gdec[3], genc[2];
  for (int l = 0; l < 3; ++l) {
    gdec[l].att  = (const float*)d_in[p++]; gdec[l].bias = (const float*)d_in[p++];
    gdec[l].bl   = (const float*)d_in[p++]; gdec[l].br   = (const float*)d_in[p++];
    gdec[l].wl   = (const float*)d_in[p++]; gdec[l].wr   = (const float*)d_in[p++];
  }
  for (int l = 0; l < 2; ++l) {
    genc[l].att  = (const float*)d_in[p++]; genc[l].bias = (const float*)d_in[p++];
    genc[l].bl   = (const float*)d_in[p++]; genc[l].br   = (const float*)d_in[p++];
    genc[l].wl   = (const float*)d_in[p++]; genc[l].wr   = (const float*)d_in[p++];
  }
  const float *b_b   = (const float*)d_in[p++], *b_w   = (const float*)d_in[p++];
  const float *d_b   = (const float*)d_in[p++], *d_w   = (const float*)d_in[p++];
  const float *dg1_b = (const float*)d_in[p++], *dg1_w = (const float*)d_in[p++];
  const float *dg2_b = (const float*)d_in[p++], *dg2_w = (const float*)d_in[p++];
  const float *mu_b  = (const float*)d_in[p++], *mu_w  = (const float*)d_in[p++];
  const float *var_b = (const float*)d_in[p++], *var_w = (const float*)d_in[p++];
  PtP pt[3];
  for (int l = 0; l < 3; ++l) {
    pt[l].attn_b    = (const float*)d_in[p++]; pt[l].attn_w    = (const float*)d_in[p++];
    pt[l].dst_w     = (const float*)d_in[p++]; pt[l].lin_in_b  = (const float*)d_in[p++];
    pt[l].lin_in_w  = (const float*)d_in[p++]; pt[l].lin_out_b = (const float*)d_in[p++];
    pt[l].lin_out_w = (const float*)d_in[p++]; pt[l].lin_w     = (const float*)d_in[p++];
    pt[l].pos_b     = (const float*)d_in[p++]; pt[l].pos_w     = (const float*)d_in[p++];
    pt[l].src_w     = (const float*)d_in[p++];
  }

  // workspace bump allocator
  char* wsp = (char*)d_ws;
  auto carve = [&](size_t bytes) -> void* {
    void* r = (void*)wsp;
    wsp += (bytes + 255) & ~((size_t)255);
    return r;
  };
  int*   srcI   = (int*)carve((size_t)NEDGE * 4);
  int*   dstI   = (int*)carve((size_t)NEDGE * 4);
  float* eb0    = (float*)carve((size_t)NEDGE * 64 * 4);   // draw/alpha_pre; GAT logits
  float* eb1    = (float*)carve((size_t)NEDGE * 64 * 4);   // alpha2
  float* hA     = (float*)carve((size_t)NN * 160 * 4);
  float* hB     = (float*)carve((size_t)NN * 160 * 4);
  float* t0     = (float*)carve((size_t)NN * 160 * 4);     // xr_node / xl
  float* t1     = (float*)carve((size_t)NN * 160 * 4);     // xr (GAT)
  float* outacc = (float*)carve((size_t)NN * 160 * 4);
  unsigned* segU = (unsigned*)carve((size_t)NN * 160 * 4);
  float* segsum = (float*)carve((size_t)NN * 160 * 4);
  float* v0     = (float*)carve((size_t)NN * 64 * 4);      // val
  float* v1     = (float*)carve((size_t)NN * 64 * 4);      // a_src
  float* v2     = (float*)carve((size_t)NN * 64 * 4);      // a_dst
  float* dflat  = (float*)carve((size_t)NN * 24 * 4);
  float* dg1b   = (float*)carve((size_t)NN * 12 * 4);
  float* dg2b   = (float*)carve((size_t)NN * 12 * 4);
  float* bfl    = (float*)carve((size_t)NN * 12 * 4);
  float* sums   = (float*)carve(256 * 4);
  float* sumsq  = (float*)carve(256 * 4);
  float* m0     = (float*)carve(256 * 4);
  float* i0     = (float*)carve(256 * 4);
  float* m1     = (float*)carve(256 * 4);
  float* i1     = (float*)carve(256 * 4);

  auto blocks = [](long long n) { return (unsigned)((n + 255) / 256); };
  auto fillf = [&](float* ptr, float v, long long n) {
    fill_f32_k<<<blocks(n), 256, 0, stream>>>(ptr, v, n);
  };
  auto fillu = [&](unsigned* ptr, unsigned v, long long n) {
    fill_u32_k<<<blocks(n), 256, 0, stream>>>(ptr, v, n);
  };
  auto gemm = [&](const float* A, const float* W, const float* b, float* Cc,
                  int M, int K, int Nc, int act) {
    int mtiles = (M + 15) / 16, ntiles = (Nc + 15) / 16;
    dim3 g((unsigned)((mtiles + WPB - 1) / WPB), (unsigned)ntiles), blk(32 * WPB);
    if (act == 0)      gemm_wmma_k<0><<<g, blk, 0, stream>>>(A, W, b, Cc, M, K, Nc, mtiles);
    else if (act == 1) gemm_wmma_k<1><<<g, blk, 0, stream>>>(A, W, b, Cc, M, K, Nc, mtiles);
    else               gemm_wmma_k<2><<<g, blk, 0, stream>>>(A, W, b, Cc, M, K, Nc, mtiles);
  };
  auto bn_inplace = [&](float* X, long long rows, int C, float* bm, float* bi) {
    colstats_k<<<C, 256, 0, stream>>>(X, (int)rows, C, sums, sumsq);
    bn_params_k<<<1, 256, 0, stream>>>(sums, sumsq, 1.0f / (float)rows, C, bm, bi);
    long long tot = rows * C;
    bn_relu_k<<<blocks(tot), 256, 0, stream>>>(X, bm, bi, tot, C);
  };

  // build src/dst with self-loops
  build_edges_k<<<blocks(NEDGE), 256, 0, stream>>>(ei, srcI, dstI);

  // -------- Point-Transformer layers --------
  auto pt_layer = [&](const float* xin, int din, int dout, const PtP& P, float* hout) {
    gemm(xin, P.lin_in_w, P.lin_in_b, t0, NN, din, din, 1);   // x = relu(x@Win + b)
    gemm(t0, P.lin_w, nullptr, v0, NN, din, dout, 0);         // val
    gemm(t0, P.src_w, nullptr, v1, NN, din, dout, 0);         // a_src
    gemm(t0, P.dst_w, nullptr, v2, NN, din, dout, 0);         // a_dst
    long long totE = (long long)NEDGE * dout;
    pt_delta_raw_k<<<blocks(totE), 256, 0, stream>>>(x, srcI, dstI, P.pos_w, P.pos_b, eb0, dout);
    colstats_k<<<dout, 256, 0, stream>>>(eb0, NEDGE, dout, sums, sumsq);
    bn_params_k<<<1, 256, 0, stream>>>(sums, sumsq, 1.0f / (float)NEDGE, dout, m0, i0);
    pt_alpha_pre_k<<<blocks(totE), 256, 0, stream>>>(eb0, v1, v2, srcI, dstI, m0, i0, dout);
    gemm(eb0, P.attn_w, P.attn_b, eb1, NEDGE, dout, dout, 0); // edge-level WMMA GEMM
    bn_inplace(eb1, NEDGE, dout, m1, i1);
    fillu(segU, ORD_NEG_INF, (long long)NN * dout);
    pt_segmax_k<<<blocks(totE), 256, 0, stream>>>(eb1, dstI, segU, dout);
    fillf(segsum, 0.f, (long long)NN * dout);
    pt_segsum_k<<<blocks(totE), 256, 0, stream>>>(eb1, dstI, segU, segsum, dout);
    fillf(outacc, 0.f, (long long)NN * dout);
    pt_scatter_k<<<blocks(totE), 256, 0, stream>>>(eb1, x, srcI, dstI, P.pos_w, P.pos_b,
                                                   m0, i0, segU, segsum, v0, outacc, dout);
    gemm(outacc, P.lin_out_w, P.lin_out_b, hout, NN, dout, dout, 1);
  };

  pt_layer(x,  7,  32, pt[0], hA);
  pt_layer(hA, 32, 64, pt[1], hB);
  pt_layer(hB, 64, 64, pt[2], hA);

  // -------- GAT layers --------
  auto gat_layer = [&](const float* xin, int din, int C, bool concat, const GatP& P, float* y) {
    int HC = HEADS * C;
    gemm(xin, P.wl, P.bl, t0, NN, din, HC, 0);   // xl
    gemm(xin, P.wr, P.br, t1, NN, din, HC, 0);   // xr
    long long totEH = (long long)NEDGE * HEADS;
    float* logits = eb0;
    gat_logits_k<<<blocks(totEH), 256, 0, stream>>>(t0, t1, P.att, srcI, dstI, logits, C);
    fillu(segU, ORD_NEG_INF, (long long)NN * HEADS);
    gat_segmax_k<<<blocks(totEH), 256, 0, stream>>>(logits, dstI, segU);
    fillf(segsum, 0.f, (long long)NN * HEADS);
    gat_segsum_k<<<blocks(totEH), 256, 0, stream>>>(logits, dstI, segU, segsum);
    fillf(outacc, 0.f, (long long)NN * HC);
    gat_scatter_k<<<blocks(totEH), 256, 0, stream>>>(logits, t0, srcI, dstI, segU, segsum,
                                                     outacc, C);
    if (concat) {
      long long tot = (long long)NN * HC;
      gat_finish_concat_k<<<blocks(tot), 256, 0, stream>>>(outacc, P.bias, y, HC);
      bn_inplace(y, NN, HC, m0, i0);
    } else {
      long long tot = (long long)NN * C;
      gat_finish_mean_k<<<blocks(tot), 256, 0, stream>>>(outacc, P.bias, y, C);
      bn_inplace(y, NN, C, m0, i0);
    }
  };

  gat_layer(hA, 64,  32, true,  genc[0], hB);   // -> N x 160
  gat_layer(hB, 160, 32, false, genc[1], hA);   // -> N x 32

  // -------- head + Sylvester flows --------
  float* out_g   = (float*)d_out;
  float* out_mu  = out_g  + (size_t)NN * 7;
  float* out_lv  = out_mu + (size_t)NN * 2;
  float* out_ldj = out_lv + (size_t)NN * 2;
  float* out_z0  = out_ldj + (size_t)NN;
  float* out_z   = out_z0 + (size_t)NN * 2;

  gemm(hA, mu_w,  mu_b,  out_mu, NN, 32, 2,  0);
  gemm(hA, var_w, var_b, out_lv, NN, 32, 2,  0);
  gemm(hA, d_w,   d_b,   dflat,  NN, 32, 24, 0);
  gemm(hA, dg1_w, dg1_b, dg1b,   NN, 32, 12, 2);   // tanh fused
  gemm(hA, dg2_w, dg2_b, dg2b,   NN, 32, 12, 2);   // tanh fused
  gemm(hA, b_w,   b_b,   bfl,    NN, 32, 12, 0);
  head_flow_k<<<blocks(NN), 256, 0, stream>>>(out_mu, out_lv, eps, dflat, dg1b, dg2b, bfl,
                                              out_z0, out_z, out_ldj);

  // -------- decoder GATs --------
  gat_layer(out_z, 2,   32, true,  gdec[0], hB);   // -> N x 160
  gat_layer(hB,    160, 32, true,  gdec[1], hA);   // -> N x 160
  gat_layer(hA,    160, 7,  false, gdec[2], out_g);// -> N x 7 (final g, bn+relu in place)
}
